// AttentionBlock_16509854286104
// MI455X (gfx1250) — matrix operand order repaired and checked by reading, not compiled, not run on hardware
//
#include <hip/hip_runtime.h>

// ---------------------------------------------------------------------------
// AttentionBlock for MI455X (gfx1250, wave32, WMMA).
// B=4, S=1024, D=1024, H=16, hd=64. All GEMMs use v_wmma_f32_16x16x32_bf16
// (bf16 inputs, f32 accumulate); softmax/LayerNorm in f32.
// d_out = [ y (4*1024*1024 f32) | attn_weights (4*1024*1024 f32) ].
// Double-buffered global_load_async_to_lds_b128 staging in the QKV GEMM
// (ASYNCcnt pipeline); global prefetch in the out-projection kernel.
// ---------------------------------------------------------------------------

typedef __attribute__((ext_vector_type(16))) __bf16 v16bf;
typedef __attribute__((ext_vector_type(8)))  float  v8f;
typedef __attribute__((vector_size(16)))     int    async_v4i;  // builtin's pointee

union BF16x16 { unsigned int u[8]; v16bf v; };

#if defined(__has_builtin)
#if __has_builtin(__builtin_amdgcn_global_load_async_to_lds_b128)
#define HAVE_ASYNC_LDS 1
#endif
#endif

#ifndef HAVE_ASYNC_LDS
#define HAVE_ASYNC_LDS 0
#endif

#if defined(__has_builtin) && __has_builtin(__builtin_amdgcn_s_wait_asynccnt)
#define WAIT_ASYNC_ALL() __builtin_amdgcn_s_wait_asynccnt(0)
#else
#define WAIT_ASYNC_ALL() asm volatile("s_wait_asynccnt 0x0" ::: "memory")
#endif

// Two-step casts: generic reinterpret to v4i*, then addrspace cast.
#define AS1_V4I(p) \
  ((__attribute__((address_space(1))) async_v4i*)((async_v4i*)(void*)(p)))
#define AS3_V4I(p) \
  ((__attribute__((address_space(3))) async_v4i*)((async_v4i*)(void*)(p)))

__device__ __forceinline__ unsigned short bf16_bits(float f) {
  union { __bf16 h; unsigned short u; } cv;
  cv.h = (__bf16)f;
  return cv.u;
}

__device__ __forceinline__ unsigned int pack_bf16(float lo, float hi) {
  return (unsigned int)bf16_bits(lo) | ((unsigned int)bf16_bits(hi) << 16);
}

// A-matrix (16x32 bf16) per-lane K index for VGPR i, lane-half h (=lane>>4):
// VGPR i -> K = 16h + 2i (pairs K,K+1), matching the B operand fills.
__device__ __forceinline__ int a_k(int i, int h) {
  return 16 * h + 2 * i;
}

// ---------------------------------------------------------------------------
// Kernel 0: f32 -> bf16 conversion (x, w_qkv, w_out).
// ---------------------------------------------------------------------------
__global__ void convert_bf16_kernel(const float* __restrict__ src,
                                    unsigned short* __restrict__ dst, int n) {
  int i = blockIdx.x * blockDim.x + threadIdx.x;
  int stride = gridDim.x * blockDim.x;
  for (; i < n; i += stride) dst[i] = bf16_bits(src[i]);
}

// ---------------------------------------------------------------------------
// Kernel 1: qkv = x @ w_qkv^T + b_qkv  (4096x1024 @ 1024x3072)
// Block tile 64(M) x 64(N), 8 waves (4x2), each wave = 16x32 output.
// Double-buffered async-to-LDS staging of 64x32 bf16 tiles (1 b128/thread per
// matrix). Scatters to q [B,H,S,64]*hd^-0.5, k [B,H,S,64], v^T [B,H,64,S].
// ---------------------------------------------------------------------------
__global__ __launch_bounds__(256) void qkv_gemm_kernel(
    const unsigned short* __restrict__ xb,    // bf16 [4096,1024]
    const unsigned short* __restrict__ wq,    // bf16 [3072,1024]
    const float* __restrict__ bias,           // [3072]
    unsigned short* __restrict__ qws,
    unsigned short* __restrict__ kws,
    unsigned short* __restrict__ vT) {
  __shared__ alignas(16) unsigned short As[2][64 * 32];
  __shared__ alignas(16) unsigned short Bs[2][64 * 32];

  const int tid = threadIdx.x, lane = tid & 31, wid = tid >> 5;
  const int lh = lane >> 4, ln = lane & 15;
  const int m_blk = blockIdx.x * 64, n_blk = blockIdx.y * 64;
  const int mw = (wid & 3) * 16, nw = (wid >> 2) * 32;

  // staging assignment: thread -> 16B (8 bf16) of the 64x32 tile
  const int r_st = tid >> 2;        // row 0..63
  const int c_st = (tid & 3) * 8;   // col 0,8,16,24

  v8f c0 = {}, c1 = {};

#if HAVE_ASYNC_LDS
  // Issue one async b128 per thread per matrix into LDS buffer `bf`.
  auto stage_async = [&](int bf, int k0) {
    const unsigned short* ga = xb + (m_blk + r_st) * 1024 + k0 + c_st;
    const unsigned short* gb = wq + (n_blk + r_st) * 1024 + k0 + c_st;
    __builtin_amdgcn_global_load_async_to_lds_b128(
        AS1_V4I(ga), AS3_V4I(&As[bf][r_st * 32 + c_st]), 0, 0);
    __builtin_amdgcn_global_load_async_to_lds_b128(
        AS1_V4I(gb), AS3_V4I(&Bs[bf][r_st * 32 + c_st]), 0, 0);
  };
  stage_async(0, 0);
#else
  const unsigned int* xg = (const unsigned int*)xb;  // row stride 512 uints
  const unsigned int* wg = (const unsigned int*)wq;
#endif

  int buf = 0;
  for (int it = 0; it < 32; ++it) {
#if HAVE_ASYNC_LDS
    WAIT_ASYNC_ALL();        // own wave's DMA done
    __syncthreads();         // all waves' DMA visible in LDS
    if (it + 1 < 32) stage_async(buf ^ 1, (it + 1) * 32);
#else
    {
      int k0 = it * 32;
      for (int t = tid; t < 1024; t += 256) {
        int rr = t >> 4, cu = t & 15;
        ((unsigned int*)As[buf])[rr * 16 + cu] =
            xg[(m_blk + rr) * 512 + (k0 >> 1) + cu];
        ((unsigned int*)Bs[buf])[rr * 16 + cu] =
            wg[(n_blk + rr) * 512 + (k0 >> 1) + cu];
      }
      __syncthreads();
    }
#endif
    const unsigned int* Au = (const unsigned int*)As[buf];
    const unsigned int* Bu = (const unsigned int*)Bs[buf];
    BF16x16 a, b0, b1;
#pragma unroll
    for (int i = 0; i < 8; ++i) {
      int ka = a_k(i, lh);
      a.u[i]  = Au[(mw + ln) * 16 + (ka >> 1)];
      int kb = lh * 16 + 2 * i;
      b0.u[i] = Bu[(nw + ln) * 16 + (kb >> 1)];
      b1.u[i] = Bu[(nw + 16 + ln) * 16 + (kb >> 1)];
    }
    c0 = __builtin_amdgcn_wmma_f32_16x16x32_bf16(false, a.v, false, b0.v,
                                                 (short)0, c0, false, false);
    c1 = __builtin_amdgcn_wmma_f32_16x16x32_bf16(false, a.v, false, b1.v,
                                                 (short)0, c1, false, false);
    __syncthreads();
    buf ^= 1;
  }

#pragma unroll
  for (int sub = 0; sub < 2; ++sub) {
    v8f c = sub ? c1 : c0;
    int n = n_blk + nw + sub * 16 + ln;
    int sel = n >> 10, cidx = n & 1023;
    int head = cidx >> 6, d = cidx & 63;
    float bv = bias[n];
#pragma unroll
    for (int r = 0; r < 8; ++r) {
      int m = m_blk + mw + r + 8 * lh;
      int b = m >> 10, s = m & 1023;
      float val = c[r] + bv;
      if (sel == 0) {
        qws[(((b * 16 + head) * 1024 + s) << 6) + d] = bf16_bits(val * 0.125f);
      } else if (sel == 1) {
        kws[(((b * 16 + head) * 1024 + s) << 6) + d] = bf16_bits(val);
      } else {
        vT[(((b * 16 + head) * 64 + d) << 10) + s] = bf16_bits(val);
      }
    }
  }
}

// ---------------------------------------------------------------------------
// Kernel 2: attention. One block per (b, 16-row q-tile), loop over 16 heads.
// Per head: scores (WMMA) -> f32 softmax in LDS -> attn-weight average
// accumulation (deterministic per-thread RMW) -> PV (WMMA) -> ctx.
// ---------------------------------------------------------------------------
__global__ __launch_bounds__(256) void attention_kernel(
    const unsigned short* __restrict__ qws,   // bf16 [B,H,S,64] (pre-scaled)
    const unsigned short* __restrict__ kws,   // bf16 [B,H,S,64]
    const unsigned short* __restrict__ vT,    // bf16 [B,H,64,S]
    float* __restrict__ ctx,                  // f32 [4096,1024]
    float* __restrict__ attnw) {              // f32 [4,1024,1024]
  __shared__ alignas(16) float sc[16 * 1024];  // score/prob tile
  __shared__ float red[256];
  __shared__ float rowstat[16];
  __shared__ float pvbuf[4 * 256];

  const int tid = threadIdx.x, lane = tid & 31, wid = tid >> 5;
  const int lh = lane >> 4, ln = lane & 15;
  const int b = blockIdx.x >> 6, qt = blockIdx.x & 63;
  const int q0 = qt * 16;
  const int g = tid >> 4, sub = tid & 15;  // softmax: 16 threads per row

  for (int h = 0; h < 16; ++h) {
    const unsigned int* qg = (const unsigned int*)(qws + (((b * 16 + h) * 1024) << 6));
    const unsigned int* kg = (const unsigned int*)(kws + (((b * 16 + h) * 1024) << 6));
    const unsigned int* vg = (const unsigned int*)(vT  + (((b * 16 + h) * 64) << 10));

    // Q fragments for kk in [0,32) and [32,64); reused across all key tiles.
    BF16x16 aq[2];
#pragma unroll
    for (int c = 0; c < 2; ++c)
#pragma unroll
      for (int i = 0; i < 8; ++i) {
        int ka = c * 32 + a_k(i, lh);
        aq[c].u[i] = qg[(q0 + ln) * 32 + (ka >> 1)];
      }

    // Scores: 64 key-tiles of 16, 8 per wave.
    for (int t = 0; t < 8; ++t) {
      int j = t * 8 + wid;
      v8f acc = {};
#pragma unroll
      for (int c = 0; c < 2; ++c) {
        BF16x16 bk;
#pragma unroll
        for (int i = 0; i < 8; ++i) {
          int kb = c * 32 + lh * 16 + 2 * i;
          bk.u[i] = kg[(j * 16 + ln) * 32 + (kb >> 1)];
        }
        acc = __builtin_amdgcn_wmma_f32_16x16x32_bf16(false, aq[c].v, false, bk.v,
                                                      (short)0, acc, false, false);
      }
#pragma unroll
      for (int r = 0; r < 8; ++r)
        sc[(r + 8 * lh) * 1024 + j * 16 + ln] = acc[r];
    }
    __syncthreads();

    // Softmax over 1024 keys per row (f32).
    {
      float mx = -3.4e38f;
      for (int cc = sub; cc < 1024; cc += 16) mx = fmaxf(mx, sc[g * 1024 + cc]);
      red[tid] = mx;
      __syncthreads();
      if (sub == 0) {
        float m2 = red[tid];
        for (int i = 1; i < 16; ++i) m2 = fmaxf(m2, red[tid + i]);
        rowstat[g] = m2;
      }
      __syncthreads();
      float rmax = rowstat[g];
      float sm = 0.f;
      for (int cc = sub; cc < 1024; cc += 16) {
        float e = __expf(sc[g * 1024 + cc] - rmax);
        sc[g * 1024 + cc] = e;
        sm += e;
      }
      red[tid] = sm;
      __syncthreads();
      if (sub == 0) {
        float s2 = 0.f;
        for (int i = 0; i < 16; ++i) s2 += red[tid + i];
        rowstat[g] = 1.0f / s2;
      }
      __syncthreads();
      float rinv = rowstat[g];
      float* aw = attnw + ((b * 1024) + q0 + g) * 1024;
      for (int cc = sub; cc < 1024; cc += 16) {
        float p = sc[g * 1024 + cc] * rinv;
        sc[g * 1024 + cc] = p;
        float upd = p * 0.0625f;  // mean over 16 heads
        aw[cc] = (h == 0) ? upd : (aw[cc] + upd);
      }
    }
    __syncthreads();

    // PV: ctx(16x64) = P(16x1024) @ V(1024x64). 4 hd-col tiles x 2 K-halves.
    {
      int nsub = wid & 3, khalf = wid >> 2;
      v8f acc = {};
      for (int step = 0; step < 16; ++step) {
        int kb0 = khalf * 512 + step * 32;
        BF16x16 ap, bv;
        const float2* scm = (const float2*)(sc + ln * 1024 + kb0);
#pragma unroll
        for (int i = 0; i < 8; ++i) {
          int ka = a_k(i, lh);
          float2 f = scm[ka >> 1];
          ap.u[i] = pack_bf16(f.x, f.y);
          int kk = lh * 16 + 2 * i;
          bv.u[i] = vg[(nsub * 16 + ln) * 512 + ((kb0 + kk) >> 1)];
        }
        acc = __builtin_amdgcn_wmma_f32_16x16x32_bf16(false, ap.v, false, bv.v,
                                                      (short)0, acc, false, false);
      }
      if (wid >= 4) {
#pragma unroll
        for (int r = 0; r < 8; ++r) pvbuf[(wid - 4) * 256 + r * 32 + lane] = acc[r];
      }
      __syncthreads();
      if (wid < 4) {
#pragma unroll
        for (int r = 0; r < 8; ++r) {
          float v2 = acc[r] + pvbuf[wid * 256 + r * 32 + lane];
          int m = q0 + r + 8 * lh;
          ctx[((b * 1024) + m) * 1024 + h * 64 + nsub * 16 + ln] = v2;
        }
      }
      __syncthreads();  // sc reused next head
    }
  }
}

// ---------------------------------------------------------------------------
// Kernel 3: out = ctx @ w_out^T + b_out; y = LayerNorm(x + out).
// One block per 16 rows; each wave owns 8 16x16 output tiles (full 1024 cols).
// ---------------------------------------------------------------------------
__global__ __launch_bounds__(256) void out_proj_ln_kernel(
    const float* __restrict__ ctx,            // f32 [4096,1024]
    const unsigned short* __restrict__ wob,   // bf16 [1024,1024]
    const float* __restrict__ b_out,
    const float* __restrict__ x,
    const float* __restrict__ ln_w,
    const float* __restrict__ ln_b,
    float* __restrict__ y) {
  __shared__ alignas(16) float rowbuf[16 * 1024];
  __shared__ float red[256];
  __shared__ float rowm[16], rowv[16];

  const int tid = threadIdx.x, lane = tid & 31, wid = tid >> 5;
  const int lh = lane >> 4, ln = lane & 15;
  const int m_blk = blockIdx.x * 16;
  const unsigned int* wg = (const unsigned int*)wob;

  v8f zero = {};
  v8f acc[8];
#pragma unroll
  for (int t = 0; t < 8; ++t) acc[t] = zero;

  for (int k0 = 0; k0 < 1024; k0 += 32) {
    BF16x16 a;
    const float2* cm = (const float2*)(ctx + (m_blk + ln) * 1024 + k0);
    if (k0 + 32 < 1024)
      __builtin_prefetch(ctx + (m_blk + ln) * 1024 + k0 + 32, 0, 1);
#pragma unroll
    for (int i = 0; i < 8; ++i) {
      int ka = a_k(i, lh);
      float2 f = cm[ka >> 1];
      a.u[i] = pack_bf16(f.x, f.y);
    }
#pragma unroll
    for (int t = 0; t < 8; ++t) {
      int j = t * 8 + wid;
      BF16x16 bw;
#pragma unroll
      for (int i = 0; i < 8; ++i) {
        int kb = lh * 16 + 2 * i;
        bw.u[i] = wg[(j * 16 + ln) * 512 + ((k0 + kb) >> 1)];
      }
      acc[t] = __builtin_amdgcn_wmma_f32_16x16x32_bf16(false, a.v, false, bw.v,
                                                       (short)0, acc[t], false, false);
    }
  }

#pragma unroll
  for (int t = 0; t < 8; ++t) {
    int j = t * 8 + wid;
    int n = j * 16 + ln;
    float bo = b_out[n];
#pragma unroll
    for (int r = 0; r < 8; ++r) {
      int m = r + 8 * lh;
      rowbuf[m * 1024 + n] = acc[t][r] + bo + x[(m_blk + m) * 1024 + n];
    }
  }
  __syncthreads();

  const int g = tid >> 4, sub = tid & 15;
  float s = 0.f, s2 = 0.f;
  for (int cc = sub; cc < 1024; cc += 16) {
    float v = rowbuf[g * 1024 + cc];
    s += v;
    s2 += v * v;
  }
  red[tid] = s;
  __syncthreads();
  if (sub == 0) {
    float a2 = 0.f;
    for (int i = 0; i < 16; ++i) a2 += red[tid + i];
    rowm[g] = a2 * (1.0f / 1024.0f);
  }
  __syncthreads();
  red[tid] = s2;
  __syncthreads();
  if (sub == 0) {
    float a2 = 0.f;
    for (int i = 0; i < 16; ++i) a2 += red[tid + i];
    float mean = rowm[g];
    rowv[g] = a2 * (1.0f / 1024.0f) - mean * mean;
  }
  __syncthreads();
  float mean = rowm[g];
  float inv = rsqrtf(rowv[g] + 1e-5f);
  for (int cc = sub; cc < 1024; cc += 16) {
    float v = rowbuf[g * 1024 + cc];
    y[(m_blk + g) * 1024 + cc] = (v - mean) * inv * ln_w[cc] + ln_b[cc];
  }
}

// ---------------------------------------------------------------------------
extern "C" void kernel_launch(void* const* d_in, const int* in_sizes, int n_in,
                              void* d_out, int out_size, void* d_ws, size_t ws_size,
                              hipStream_t stream) {
  (void)in_sizes; (void)n_in; (void)out_size; (void)ws_size;

  const float* x     = (const float*)d_in[0];
  const float* w_qkv = (const float*)d_in[1];
  const float* b_qkv = (const float*)d_in[2];
  const float* w_out = (const float*)d_in[3];
  const float* b_out = (const float*)d_in[4];
  const float* ln_w  = (const float*)d_in[5];
  const float* ln_b  = (const float*)d_in[6];

  float* y     = (float*)d_out;
  float* attnw = y + (size_t)4 * 1024 * 1024;

  char* ws = (char*)d_ws;
  unsigned short* xb    = (unsigned short*)(ws);                         // 8 MB
  unsigned short* wqkvb = (unsigned short*)(ws + ((size_t)8  << 20));    // 6 MB
  unsigned short* woutb = (unsigned short*)(ws + ((size_t)14 << 20));    // 2 MB
  unsigned short* qws   = (unsigned short*)(ws + ((size_t)16 << 20));    // 8 MB
  unsigned short* kws   = (unsigned short*)(ws + ((size_t)24 << 20));    // 8 MB
  unsigned short* vT    = (unsigned short*)(ws + ((size_t)32 << 20));    // 8 MB
  float*          ctx   = (float*)(ws + ((size_t)40 << 20));             // 16 MB

  convert_bf16_kernel<<<2048, 256, 0, stream>>>(x, xb, 4 * 1024 * 1024);
  convert_bf16_kernel<<<2048, 256, 0, stream>>>(w_qkv, wqkvb, 3 * 1024 * 1024);
  convert_bf16_kernel<<<1024, 256, 0, stream>>>(w_out, woutb, 1024 * 1024);

  qkv_gemm_kernel<<<dim3(64, 48), 256, 0, stream>>>(xb, wqkvb, b_qkv, qws, kws, vT);
  attention_kernel<<<256, 256, 0, stream>>>(qws, kws, vT, ctx, attnw);
  out_proj_ln_kernel<<<256, 256, 0, stream>>>(ctx, woutb, b_out, x, ln_w, ln_b, y);
}
